// AttentionLSTM_33200097198958
// MI455X (gfx1250) — matrix operand order repaired and checked by reading, not compiled, not run on hardware
//
#include <hip/hip_runtime.h>

// Problem constants (match reference)
#define B_SZ        512
#define T_LEN       80
#define IN_CH       512
#define HID         512
#define NUM_CLASSES 6625
#define STEPS       25
#define GATES4      (4 * HID)
#define WIH_LD      (IN_CH + NUM_CLASSES)   // 7137
#define XH_LD       (IN_CH + HID)           // 1024

typedef __attribute__((ext_vector_type(16))) __bf16 v16bf;
typedef __attribute__((ext_vector_type(8)))  float  v8f;

__device__ __forceinline__ unsigned short f32_to_bf16(float f) {
  unsigned int u = __float_as_uint(f);
  u += 0x7FFFu + ((u >> 16) & 1u);          // round-to-nearest-even
  return (unsigned short)(u >> 16);
}

__device__ __forceinline__ float sigmoid_f(float x) {
  return 1.0f / (1.0f + __expf(-x));
}

// ---------------------------------------------------------------------------
// fp32 -> bf16 bulk convert
// ---------------------------------------------------------------------------
__global__ void __launch_bounds__(256)
cvt_f32_bf16(const float* __restrict__ src, unsigned short* __restrict__ dst, int n) {
  int i = blockIdx.x * 256 + threadIdx.x;
  if (i < n) dst[i] = f32_to_bf16(src[i]);
}

// Pack [w_ih[:, :IN_CH] | w_hh] -> wcat [4H, IN_CH+HID] bf16
__global__ void __launch_bounds__(256)
pack_wcat(const float* __restrict__ w_ih, const float* __restrict__ w_hh,
          unsigned short* __restrict__ wcat) {
  int i = blockIdx.x * 256 + threadIdx.x;
  if (i >= GATES4 * XH_LD) return;
  int n = i / XH_LD;
  int k = i - n * XH_LD;
  float v = (k < IN_CH) ? w_ih[(size_t)n * WIH_LD + k]
                        : w_hh[(size_t)n * HID + (k - IN_CH)];
  wcat[i] = f32_to_bf16(v);
}

// Zero LSTM state: c (fp32), c_bf16, and the whole [context|h] concat buffer
__global__ void __launch_bounds__(256)
init_state(float* __restrict__ c, unsigned short* __restrict__ c_bf,
           unsigned short* __restrict__ xh) {
  int i = blockIdx.x * 256 + threadIdx.x;
  if (i < B_SZ * HID) { c[i] = 0.0f; c_bf[i] = 0; }
  if (i < B_SZ * XH_LD) xh[i] = 0;
}

// ---------------------------------------------------------------------------
// Fragment layout helpers (CDNA5 wave32, V_WMMA_F32_16X16X32_BF16):
//  A (16x32 bf16): lane l: M = l&15; dword offsets = k/2 + (l>>4)*4 (+0..3)
//                  and +8 for K+16 half.
//  B (32x16 bf16): lane l: N = l&15; same K order as A:
//                  dword offsets = k/2 + (l>>4)*4 (+0..3), +8 for K+16 half.
//  C/D (16x16 f32): lane l: N = l&15; VGPR r: M = r + 8*(l>>4).
// ---------------------------------------------------------------------------

// Fine-grained GEMM: 1 accumulator tile / wave, block tile 32x64.
// Best for the small per-step GEMMs (latency / occupancy bound).
__global__ void __launch_bounds__(256)
wmma_gemm_bf16(const unsigned short* __restrict__ A, int lda,
               const unsigned short* __restrict__ W, int ldb,
               float* __restrict__ C, int ldc,
               const float* __restrict__ bias,
               int M, int N, int K)
{
  const int lane = threadIdx.x & 31;
  const int wave = threadIdx.x >> 5;
  const int wm = wave >> 2;           // 0..1
  const int wn = wave & 3;            // 0..3
  const int row0 = blockIdx.y * 32 + wm * 16;
  const int col0 = blockIdx.x * 64 + wn * 16;
  if (row0 >= M) return;

  const int l15 = lane & 15;
  const int hi  = lane >> 4;
  const int rowA = row0 + l15;
  const int colB = col0 + l15;
  const bool colOK = (colB < N);

  const unsigned int* __restrict__ pa =
      (const unsigned int*)(A + (size_t)rowA * lda);
  const unsigned int* __restrict__ pb =
      (const unsigned int*)(W + (size_t)(colOK ? colB : 0) * ldb);

  v8f acc = {};
  for (int kk = 0; kk < K; kk += 32) {
    union { v16bf v; unsigned int u[8]; } af, bfg;
    const int ka = (kk >> 1) + hi * 4;
    const int kb = (kk >> 1) + hi * 8;
#pragma unroll
    for (int v = 0; v < 4; ++v) {
      af.u[v]     = pa[ka + v];
      af.u[4 + v] = pa[ka + 8 + v];
    }
#pragma unroll
    for (int v = 0; v < 8; ++v) bfg.u[v] = pb[kb - hi * 4 + (v & 3) + (v >> 2) * 8];

    if (kk + 32 < K) {
      __builtin_prefetch(pa + ka + 16, 0, 3);
      __builtin_prefetch(pb + kb + 16, 0, 3);
    }
    acc = __builtin_amdgcn_wmma_f32_16x16x32_bf16(
        false, af.v, false, bfg.v, (short)0, acc, false, false);
  }

  if (colOK) {
    const float badd = bias ? bias[colB] : 0.0f;
#pragma unroll
    for (int r = 0; r < 8; ++r) {
      const int outRow = row0 + hi * 8 + r;
      if (outRow < M)
        C[(size_t)outRow * ldc + colB] = acc[r] + badd;
    }
  }
}

// Register-blocked GEMM: each wave owns a 2(M) x 4(N) grid of 16x16
// accumulators (32x64 output / wave). 8 waves tiled 4(M) x 2(N) -> block
// tile 128x128. Per K-chunk: 2 A-frags + 4 B-frags feed 8 WMMAs, giving
// 4x A-reuse / 2x B-reuse and ~2.7x less L2 fragment traffic than the
// fine kernel. Used for the big compute-bound GEMMs (proj, generator).
__global__ void __launch_bounds__(256)
wmma_gemm_bf16_rb(const unsigned short* __restrict__ A, int lda,
                  const unsigned short* __restrict__ W, int ldb,
                  float* __restrict__ C, int ldc,
                  const float* __restrict__ bias,
                  int M, int N, int K)
{
  const int lane = threadIdx.x & 31;
  const int wave = threadIdx.x >> 5;
  const int wm = wave >> 1;           // 0..3
  const int wn = wave & 1;            // 0..1
  const int row0 = blockIdx.y * 128 + wm * 32;   // 2 M-tiles
  const int col0 = blockIdx.x * 128 + wn * 64;   // 4 N-tiles
  if (row0 >= M) return;

  const int l15 = lane & 15;
  const int hi  = lane >> 4;

  const unsigned int* __restrict__ pa[2];
  const unsigned int* __restrict__ pb[4];
#pragma unroll
  for (int i = 0; i < 2; ++i) {
    int r = row0 + i * 16 + l15;
    pa[i] = (const unsigned int*)(A + (size_t)(r < M ? r : 0) * lda);
  }
#pragma unroll
  for (int j = 0; j < 4; ++j) {
    int cB = col0 + j * 16 + l15;
    pb[j] = (const unsigned int*)(W + (size_t)(cB < N ? cB : 0) * ldb);
  }

  v8f acc[2][4] = {};
  for (int kk = 0; kk < K; kk += 32) {
    union { v16bf v; unsigned int u[8]; } af[2], bfr[4];
    const int ka = (kk >> 1) + hi * 4;
    const int kb = (kk >> 1) + hi * 8;
#pragma unroll
    for (int i = 0; i < 2; ++i) {
#pragma unroll
      for (int v = 0; v < 4; ++v) {
        af[i].u[v]     = pa[i][ka + v];
        af[i].u[4 + v] = pa[i][ka + 8 + v];
      }
    }
#pragma unroll
    for (int j = 0; j < 4; ++j) {
#pragma unroll
      for (int v = 0; v < 8; ++v) bfr[j].u[v] = pb[j][kb - hi * 4 + (v & 3) + (v >> 2) * 8];
    }
    if (kk + 32 < K) {
#pragma unroll
      for (int i = 0; i < 2; ++i) __builtin_prefetch(pa[i] + ka + 16, 0, 3);
#pragma unroll
      for (int j = 0; j < 4; ++j) __builtin_prefetch(pb[j] + kb + 16, 0, 3);
    }
#pragma unroll
    for (int i = 0; i < 2; ++i)
#pragma unroll
      for (int j = 0; j < 4; ++j)
        acc[i][j] = __builtin_amdgcn_wmma_f32_16x16x32_bf16(
            false, af[i].v, false, bfr[j].v, (short)0, acc[i][j],
            false, false);
  }

#pragma unroll
  for (int j = 0; j < 4; ++j) {
    const int colB = col0 + j * 16 + l15;
    if (colB >= N) continue;
    const float badd = bias ? bias[colB] : 0.0f;
#pragma unroll
    for (int i = 0; i < 2; ++i) {
#pragma unroll
      for (int r = 0; r < 8; ++r) {
        const int outRow = row0 + i * 16 + hi * 8 + r;
        if (outRow < M)
          C[(size_t)outRow * ldc + colB] = acc[i][j][r] + badd;
      }
    }
  }
}

// ---------------------------------------------------------------------------
// Fused attention step (one block per batch row b):
//   e[t]     = sum_h tanh(proj[b,t,h] + s[b,h]) * score_w[h]
//   alpha    = softmax_t(e)
//   context  = sum_t alpha[t] * inputs[b,t,:]
// Writes context fp32 and bf16 into xh[:, 0:IN_CH].
// proj + inputs (168 MB) are L2-resident across the 25 steps.
// ---------------------------------------------------------------------------
__global__ void __launch_bounds__(256)
attention_step(const float* __restrict__ proj,      // [B,T,H]
               const float* __restrict__ s,         // [B,H]
               const float* __restrict__ score_w,   // [H]
               const float* __restrict__ inputs,    // [B,T,D]
               float* __restrict__ context,         // [B,D]
               unsigned short* __restrict__ xh)     // [B, XH_LD]
{
  __shared__ float s_sh[HID];
  __shared__ float w_sh[HID];
  __shared__ float e_sh[T_LEN];
  __shared__ float p_sh[T_LEN];

  const int b   = blockIdx.x;
  const int tid = threadIdx.x;
  const int lane = tid & 31;
  const int wave = tid >> 5;

  for (int i = tid; i < HID; i += 256) {
    s_sh[i] = s[(size_t)b * HID + i];
    w_sh[i] = score_w[i];
  }
  __syncthreads();

  const float* projb = proj + (size_t)b * T_LEN * HID;
  for (int t = wave; t < T_LEN; t += 8) {               // 8 waves cover T=80
    const float* pr = projb + (size_t)t * HID;
    float part = 0.0f;
    for (int h = lane; h < HID; h += 32)
      part += tanhf(pr[h] + s_sh[h]) * w_sh[h];
#pragma unroll
    for (int off = 16; off > 0; off >>= 1)              // wave32 reduction
      part += __shfl_down(part, off, 32);
    if (lane == 0) e_sh[t] = part;
  }
  __syncthreads();

  float m = -3.0e38f;
  for (int t = 0; t < T_LEN; ++t) m = fmaxf(m, e_sh[t]);
  if (tid < T_LEN) p_sh[tid] = __expf(e_sh[tid] - m);
  __syncthreads();
  float sum = 0.0f;
  for (int t = 0; t < T_LEN; ++t) sum += p_sh[t];
  const float inv = 1.0f / sum;

  const float* inb = inputs + (size_t)b * T_LEN * IN_CH;
  for (int d = tid; d < IN_CH; d += 256) {
    float acc = 0.0f;
    for (int t = 0; t < T_LEN; ++t)
      acc += p_sh[t] * inb[(size_t)t * IN_CH + d];
    acc *= inv;
    context[(size_t)b * IN_CH + d] = acc;
    xh[(size_t)b * XH_LD + d] = f32_to_bf16(acc);       // bf16 concat, ctx half
  }
}

// ---------------------------------------------------------------------------
// LSTM cell elementwise. Adds biases + one-hot column gather of w_ih, then
//   c' = sig(f)*c + sig(i)*tanh(g);  h' = sig(o)*tanh(c')
// Writes c (fp32), c_bf16 (query for next s-GEMM), xh h-half, hs[b,step,:].
// ---------------------------------------------------------------------------
__global__ void __launch_bounds__(256)
lstm_cell(const float* __restrict__ gates,     // [B, 4H]
          const float* __restrict__ b_ih,      // [4H]
          const float* __restrict__ b_hh,      // [4H]
          const float* __restrict__ w_ih,      // [4H, WIH_LD] fp32
          const int*   __restrict__ targets,   // [B, STEPS]
          int step,
          float* __restrict__ c,               // [B,H] in/out
          unsigned short* __restrict__ c_bf,   // [B,H]
          unsigned short* __restrict__ xh,     // [B, XH_LD] (h half)
          unsigned short* __restrict__ hs)     // [B, STEPS, H]
{
  const int idx = blockIdx.x * 256 + threadIdx.x;       // over B*H
  if (idx >= B_SZ * HID) return;
  const int b = idx / HID;
  const int j = idx - b * HID;
  const int tgt = targets[b * STEPS + step];
  const size_t col = (size_t)IN_CH + (size_t)tgt;
  const size_t gb = (size_t)b * GATES4;

  const float gi = gates[gb + j]
                 + b_ih[j]           + b_hh[j]
                 + w_ih[(size_t)j * WIH_LD + col];
  const float gf = gates[gb + HID + j]
                 + b_ih[HID + j]     + b_hh[HID + j]
                 + w_ih[(size_t)(HID + j) * WIH_LD + col];
  const float gg = gates[gb + 2 * HID + j]
                 + b_ih[2 * HID + j] + b_hh[2 * HID + j]
                 + w_ih[(size_t)(2 * HID + j) * WIH_LD + col];
  const float go = gates[gb + 3 * HID + j]
                 + b_ih[3 * HID + j] + b_hh[3 * HID + j]
                 + w_ih[(size_t)(3 * HID + j) * WIH_LD + col];

  const float cn = sigmoid_f(gf) * c[idx] + sigmoid_f(gi) * tanhf(gg);
  const float hn = sigmoid_f(go) * tanhf(cn);

  c[idx]    = cn;
  c_bf[idx] = f32_to_bf16(cn);
  xh[(size_t)b * XH_LD + IN_CH + j] = f32_to_bf16(hn);
  hs[((size_t)b * STEPS + step) * HID + j] = f32_to_bf16(hn);
}

// ---------------------------------------------------------------------------
extern "C" void kernel_launch(void* const* d_in, const int* in_sizes, int n_in,
                              void* d_out, int out_size, void* d_ws, size_t ws_size,
                              hipStream_t stream) {
  const float* inputs  = (const float*)d_in[0];   // [B,T,D]
  const int*   targets = (const int*)  d_in[1];   // [B,STEPS]
  const float* i2h_w   = (const float*)d_in[2];   // [H,D]
  const float* h2h_w   = (const float*)d_in[3];   // [H,H]
  const float* h2h_b   = (const float*)d_in[4];   // [H]
  const float* score_w = (const float*)d_in[5];   // [1,H]
  const float* w_ih    = (const float*)d_in[6];   // [4H, D+C]
  const float* w_hh    = (const float*)d_in[7];   // [4H, H]
  const float* b_ih    = (const float*)d_in[8];   // [4H]
  const float* b_hh    = (const float*)d_in[9];   // [4H]
  const float* gen_w   = (const float*)d_in[10];  // [C,H]
  const float* gen_b   = (const float*)d_in[11];  // [C]
  float* out = (float*)d_out;                     // [B,STEPS,C]

  // ---- carve workspace ----
  char* ws = (char*)d_ws;
  size_t off = 0;
  auto carve = [&](size_t bytes) -> void* {
    void* p = ws + off;
    off = (off + bytes + 255) & ~(size_t)255;
    return p;
  };
  unsigned short* inputs_bf = (unsigned short*)carve((size_t)B_SZ * T_LEN * IN_CH * 2);
  float*          proj      = (float*)         carve((size_t)B_SZ * T_LEN * HID * 4);
  unsigned short* i2h_bf    = (unsigned short*)carve((size_t)HID * IN_CH * 2);
  unsigned short* h2h_bf    = (unsigned short*)carve((size_t)HID * HID * 2);
  unsigned short* genw_bf   = (unsigned short*)carve((size_t)NUM_CLASSES * HID * 2);
  unsigned short* wcat_bf   = (unsigned short*)carve((size_t)GATES4 * XH_LD * 2);
  float*          s_buf     = (float*)         carve((size_t)B_SZ * HID * 4);
  float*          c_state   = (float*)         carve((size_t)B_SZ * HID * 4);
  unsigned short* c_bf      = (unsigned short*)carve((size_t)B_SZ * HID * 2);
  unsigned short* xh_bf     = (unsigned short*)carve((size_t)B_SZ * XH_LD * 2);
  float*          gates     = (float*)         carve((size_t)B_SZ * GATES4 * 4);
  float*          ctx       = (float*)         carve((size_t)B_SZ * IN_CH * 4);
  unsigned short* hs_bf     = (unsigned short*)carve((size_t)B_SZ * STEPS * HID * 2);
  (void)ws_size; (void)in_sizes; (void)n_in; (void)out_size;

  auto cvt = [&](const float* src, unsigned short* dst, int n) {
    cvt_f32_bf16<<<(n + 255) / 256, 256, 0, stream>>>(src, dst, n);
  };
  // small-GEMM path: 32x64 block tile, max parallelism
  auto gemm_fine = [&](const unsigned short* A, int lda, const unsigned short* W,
                       int ldb, float* C, int ldc, const float* bias,
                       int M, int N, int K) {
    dim3 g((N + 63) / 64, (M + 31) / 32);
    wmma_gemm_bf16<<<g, 256, 0, stream>>>(A, lda, W, ldb, C, ldc, bias, M, N, K);
  };
  // big-GEMM path: 128x128 block tile, register-blocked 2x4 tiles/wave
  auto gemm_big = [&](const unsigned short* A, int lda, const unsigned short* W,
                      int ldb, float* C, int ldc, const float* bias,
                      int M, int N, int K) {
    dim3 g((N + 127) / 128, (M + 127) / 128);
    wmma_gemm_bf16_rb<<<g, 256, 0, stream>>>(A, lda, W, ldb, C, ldc, bias, M, N, K);
  };

  // ---- one-time prep ----
  cvt(inputs, inputs_bf, B_SZ * T_LEN * IN_CH);
  cvt(i2h_w,  i2h_bf,  HID * IN_CH);
  cvt(h2h_w,  h2h_bf,  HID * HID);
  cvt(gen_w,  genw_bf, NUM_CLASSES * HID);
  pack_wcat<<<(GATES4 * XH_LD + 255) / 256, 256, 0, stream>>>(w_ih, w_hh, wcat_bf);
  init_state<<<(B_SZ * XH_LD + 255) / 256, 256, 0, stream>>>(c_state, c_bf, xh_bf);

  // proj[B*T, H] = inputs @ i2h_w^T  (step-invariant, hoisted)
  gemm_big(inputs_bf, IN_CH, i2h_bf, IN_CH, proj, HID, nullptr,
           B_SZ * T_LEN, HID, IN_CH);

  // ---- decode steps ----
  for (int step = 0; step < STEPS; ++step) {
    // s = c @ h2h_w^T + h2h_b
    gemm_fine(c_bf, HID, h2h_bf, HID, s_buf, HID, h2h_b, B_SZ, HID, HID);
    // attention -> context (+ bf16 into xh[:, :D])
    attention_step<<<B_SZ, 256, 0, stream>>>(proj, s_buf, score_w, inputs,
                                             ctx, xh_bf);
    // gates = [context|h] @ [w_ih_ctx|w_hh]^T   (biases + one-hot in epilogue)
    gemm_fine(xh_bf, XH_LD, wcat_bf, XH_LD, gates, GATES4, nullptr,
              B_SZ, GATES4, XH_LD);
    // LSTM cell elementwise
    lstm_cell<<<(B_SZ * HID + 255) / 256, 256, 0, stream>>>(
        gates, b_ih, b_hh, w_ih, targets, step,
        c_state, c_bf, xh_bf, hs_bf);
  }

  // probs[B*S, C] = hs @ gen_w^T + gen_b   (row b*S+s matches [B,S,C])
  gemm_big(hs_bf, HID, genw_bf, HID, out, NUM_CLASSES, gen_b,
           B_SZ * STEPS, NUM_CLASSES, HID);
}
